// NTXentLoss_36515811951305
// MI455X (gfx1250) — matrix operand order repaired and checked by reading, not compiled, not run on hardware
//
#include <hip/hip_runtime.h>

typedef __attribute__((ext_vector_type(16))) _Float16 v16h;
typedef __attribute__((ext_vector_type(8)))  float    v8f;
typedef unsigned int u32x4 __attribute__((ext_vector_type(4)));
typedef int          i32x4 __attribute__((ext_vector_type(4)));
typedef int          i32x8 __attribute__((ext_vector_type(8)));

#define N_ROWS   4096
#define DIM      256
#define TWO_N    8192
#define NJB      64        // column blocks of 128
#define INV_T    2.0f      // 1 / temperature(0.5)

union FragH { uint4 u[2]; v16h h; };
union Oct   { uint4 u; _Float16 h[8]; };

__device__ __forceinline__ float wave_sum32(float v) {
    #pragma unroll
    for (int m = 16; m >= 1; m >>= 1) v += __shfl_xor(v, m, 32);
    return v;
}

// ---------------------------------------------------------------- normalize
// one wave (32 lanes) per row; each lane handles 8 contiguous floats
__global__ __launch_bounds__(256) void nrm_kernel(
        const float* __restrict__ zi, const float* __restrict__ zj,
        _Float16* __restrict__ R) {
    int wid  = (blockIdx.x * 256 + threadIdx.x) >> 5;   // row 0..8191
    int lane = threadIdx.x & 31;
    const float* src = (wid < N_ROWS) ? (zi + (size_t)wid * DIM)
                                      : (zj + (size_t)(wid - N_ROWS) * DIM);
    const float4* s4 = (const float4*)src;
    float4 p0 = s4[lane * 2];
    float4 p1 = s4[lane * 2 + 1];
    float ss = p0.x*p0.x + p0.y*p0.y + p0.z*p0.z + p0.w*p0.w
             + p1.x*p1.x + p1.y*p1.y + p1.z*p1.z + p1.w*p1.w;
    ss = wave_sum32(ss);
    float inv = 1.0f / fmaxf(sqrtf(ss), 1e-12f);
    Oct o;
    o.h[0]=(_Float16)(p0.x*inv); o.h[1]=(_Float16)(p0.y*inv);
    o.h[2]=(_Float16)(p0.z*inv); o.h[3]=(_Float16)(p0.w*inv);
    o.h[4]=(_Float16)(p1.x*inv); o.h[5]=(_Float16)(p1.y*inv);
    o.h[6]=(_Float16)(p1.z*inv); o.h[7]=(_Float16)(p1.w*inv);
    ((uint4*)(R + (size_t)wid * DIM))[lane] = o.u;
}

// --------------------------------------------------- Gram tile + exp + Σcols
// grid (64,64): block (ib,jb) covers rows ib*128..+128, cols jb*128..+128.
// 8 waves; wave w owns rows w*16..+16. B tile (64 KB, contiguous in global)
// is DMA'd into LDS by the Tensor Data Mover, programmed by wave 0.
__global__ __launch_bounds__(256) void simexp_kernel(
        const _Float16* __restrict__ R, float* __restrict__ partial) {
    __shared__ _Float16 sB[128 * DIM];                  // 64 KB, LDS offset 0
    const int ib = blockIdx.x, jb = blockIdx.y;

    // ---- TDM: DMA rows jb*128..+128 of R (contiguous 64 KB) into sB ----
    if (threadIdx.x < 32) {                             // wave 0 programs the D#
        unsigned long long ga =
            (unsigned long long)(R + (size_t)jb * 128 * DIM);
        // D# group0: count=1 | lds_addr=0 | global_addr[56:0] | type=2
        u32x4 g0 = { 1u,
                     0u,
                     (unsigned)ga,
                     (unsigned)((ga >> 32) & 0x1FFFFFFull) | 0x80000000u };
        // D# group1: data_size=3 (8B); tensor_dim0=tile_dim0=8192 elements;
        // tensor_dim1=1, tile_dim1=1, stride0=8192 (one 64 KB row)
        i32x8 g1 = { 0x00030000,        // workgroup_mask=0, data_size=3
                     0x20000000,        // tensor_dim0[15:0]=8192 in [31:16]
                     0x00010000,        // tensor_dim0[31:16]=0, tensor_dim1=1
                     0x20000000,        // tile_dim0=8192 in [31:16]
                     0x00000001,        // tile_dim1=1, tile_dim2=0
                     8192,              // tensor_dim0_stride[31:0]
                     0, 0 };
        i32x4 z4 = { 0, 0, 0, 0 };
#if defined(__clang_major__) && (__clang_major__ >= 23)
        i32x8 z8 = { 0, 0, 0, 0, 0, 0, 0, 0 };
        __builtin_amdgcn_tensor_load_to_lds(g0, g1, z4, z4, z8, 0);
#else
        __builtin_amdgcn_tensor_load_to_lds(g0, g1, z4, z4, 0);
#endif
        __builtin_amdgcn_s_wait_tensorcnt(0);
    }
    // The TDM intrinsic doesn't reference sB, so LLVM's no-escape analysis
    // would otherwise prove sB is never written and fold the ds_loads to
    // undef. Escape the pointer + clobber memory so the DMA's writes are
    // observable to the optimizer.
    asm volatile("" :: "v"(&sB[0]) : "memory");
    __syncthreads();

    const int wave = threadIdx.x >> 5;
    const int lane = threadIdx.x & 31;
    const int g    = lane >> 4;        // K half-group
    const int m    = lane & 15;        // row/col within 16

    v8f acc[8];
    #pragma unroll
    for (int t = 0; t < 8; ++t) acc[t] = v8f{0.f,0.f,0.f,0.f,0.f,0.f,0.f,0.f};

    const uint4* sB4 = (const uint4*)sB;
    // A fragments: row (ib*128 + wave*16 + m), 32 uint4 per row, straight
    // from global (R is L2-resident; each wave reads its 16 rows once)
    const uint4* a4 = (const uint4*)(R + (size_t)(ib * 128 + wave * 16 + m) * DIM);

    #pragma unroll
    for (int kk = 0; kk < 8; ++kk) {                    // K = 256 in steps of 32
        FragH fa;                                       // 16-bit A 16x32 layout
        fa.u[0] = a4[kk * 4 + g];                       // K = kk*32 + g*8 .. +8
        fa.u[1] = a4[kk * 4 + 2 + g];                   // K = kk*32 + 16 + g*8
        // batch-load all 8 B fragments first so the 16 ds_load_b128 get
        // clause'd and the 8 WMMAs issue back-to-back behind one wait
        FragH fb[8];
        #pragma unroll
        for (int jt = 0; jt < 8; ++jt) {
            const uint4* b4 = sB4 + (jt * 16 + m) * 32; // col jt*16+m of tile
            fb[jt].u[0] = b4[kk * 4 + g];               // K = kk*32 + g*8 .. +8
            fb[jt].u[1] = b4[kk * 4 + 2 + g];           // K = kk*32 + 16 + g*8
        }
        #pragma unroll
        for (int jt = 0; jt < 8; ++jt)
            acc[jt] = __builtin_amdgcn_wmma_f32_16x16x32_f16(
                false, fa.h, false, fb[jt].h, (short)0, acc[jt], false, false);
    }

    // acc[jt][v] = sim row (8g+v) x col (jt*16+m); exp then sum over columns
    float rsum[8] = {0.f,0.f,0.f,0.f,0.f,0.f,0.f,0.f};
    #pragma unroll
    for (int jt = 0; jt < 8; ++jt)
        #pragma unroll
        for (int v = 0; v < 8; ++v)
            rsum[v] += __expf(acc[jt][v] * INV_T);
    #pragma unroll
    for (int v = 0; v < 8; ++v) {
        #pragma unroll
        for (int msk = 8; msk >= 1; msk >>= 1)          // reduce within 16-lane half
            rsum[v] += __shfl_xor(rsum[v], msk, 32);
    }
    if (m == 0) {
        int row = ib * 128 + wave * 16 + g * 8;
        #pragma unroll
        for (int v = 0; v < 8; ++v)
            partial[(size_t)jb * TWO_N + row + v] = rsum[v];
    }
}

// ------------------------------------------------------------- per-row loss
// one wave per row: fixed-order sum of 64 partials, minus diagonal exp,
// minus positive logit.
__global__ __launch_bounds__(256) void rowloss_kernel(
        const _Float16* __restrict__ R, const float* __restrict__ partial,
        float* __restrict__ rowloss) {
    int row  = (blockIdx.x * 256 + threadIdx.x) >> 5;
    int lane = threadIdx.x & 31;
    int partner = (row + N_ROWS) & (TWO_N - 1);

    float ps = partial[(size_t)(2 * lane)     * TWO_N + row]
             + partial[(size_t)(2 * lane + 1) * TWO_N + row];

    Oct ra, rb;
    ra.u = ((const uint4*)(R + (size_t)row     * DIM))[lane];
    rb.u = ((const uint4*)(R + (size_t)partner * DIM))[lane];
    float sd = 0.f, pd = 0.f;
    #pragma unroll
    for (int i = 0; i < 8; ++i) {
        float a = (float)ra.h[i], b = (float)rb.h[i];
        sd += a * a;
        pd += a * b;
    }
    ps = wave_sum32(ps);
    sd = wave_sum32(sd);
    pd = wave_sum32(pd);
    if (lane == 0)
        rowloss[row] = logf(ps - __expf(sd * INV_T)) - pd * INV_T;
}

// ----------------------------------------------------------------- mean
__global__ __launch_bounds__(256) void meanloss_kernel(
        const float* __restrict__ rowloss, float* __restrict__ out) {
    __shared__ float s[8];
    float v = 0.f;
    for (int i = threadIdx.x; i < TWO_N; i += 256) v += rowloss[i];
    v = wave_sum32(v);
    if ((threadIdx.x & 31) == 0) s[threadIdx.x >> 5] = v;
    __syncthreads();
    if (threadIdx.x == 0) {
        float t = 0.f;
        #pragma unroll
        for (int i = 0; i < 8; ++i) t += s[i];
        out[0] = t / (float)TWO_N;
    }
}

extern "C" void kernel_launch(void* const* d_in, const int* in_sizes, int n_in,
                              void* d_out, int out_size, void* d_ws, size_t ws_size,
                              hipStream_t stream) {
    const float* zi = (const float*)d_in[0];
    const float* zj = (const float*)d_in[1];
    float* out = (float*)d_out;

    char* ws = (char*)d_ws;
    _Float16* R       = (_Float16*)ws;                           // 4 MB
    float*    partial = (float*)(ws + (size_t)4 * 1024 * 1024);  // 2 MB (64 x 8192)
    float*    rowloss = (float*)(ws + (size_t)6 * 1024 * 1024);  // 32 KB

    nrm_kernel<<<TWO_N / 8, 256, 0, stream>>>(zi, zj, R);
    simexp_kernel<<<dim3(NJB, NJB), 256, 0, stream>>>(R, partial);
    rowloss_kernel<<<TWO_N / 8, 256, 0, stream>>>(R, partial, rowloss);
    meanloss_kernel<<<1, 256, 0, stream>>>(rowloss, out);
}
